// BinLinear_20126216749918
// MI455X (gfx1250) — compile-verified, matrix operand order checked by reading, not run on hardware
//
#include <hip/hip_runtime.h>

// ---------------------------------------------------------------------------
// BinLinear: y = x @ sign(W)^T + sign(b)   (M=8192, K=4096, N=4096, fp32)
//
// - sign(W) exact in f16; x split into hi+lo f16 terms -> two
//   v_wmma_f32_16x16x32_f16 per tile with f32 accumulation => ~fp32 accuracy
//   at f16 matrix-pipe throughput.
// - 64x64 per-wave tile (4x4 of 16x16): 32 WMMAs per 24 ds_load_b128 per
//   K-step, so the matrix pipe (not LDS read bandwidth) is the bottleneck.
// - Split + binarization fused into the LDS fill; fragment loads are pure
//   ds_load_b128 (zero consume-side VALU).
// - Register-staged prefetch of the next K-tile issued before compute.
// ---------------------------------------------------------------------------

typedef _Float16 v16h __attribute__((ext_vector_type(16)));
typedef _Float16 v8h  __attribute__((ext_vector_type(8)));
typedef _Float16 v4h  __attribute__((ext_vector_type(4)));
typedef float    v8f  __attribute__((ext_vector_type(8)));
typedef float    v4f  __attribute__((ext_vector_type(4)));

#define BM 128
#define BN 256
#define BK 32
#define HS 40   // halves per LDS row (32 data + 8 pad) -> 80B stride, 16B aligned

__device__ __forceinline__ float signf(float v) {
    return v > 0.0f ? 1.0f : (v < 0.0f ? -1.0f : 0.0f);
}

// v16h element i holds K = kb + i + (i>=8 ? 8 : 0)  (ISA 16-bit A layout):
// two K-contiguous 16-byte chunks at +kb and +kb+16.
__device__ __forceinline__ v16h load_frag(const _Float16* p) {
    v8h a = *(const v8h*)(p);
    v8h b = *(const v8h*)(p + 16);
    return __builtin_shufflevector(a, b, 0, 1, 2, 3, 4, 5, 6, 7,
                                         8, 9, 10, 11, 12, 13, 14, 15);
}

__global__ __launch_bounds__(256)
void binlinear_wmma_f16split(const float* __restrict__ x,
                             const float* __restrict__ w,
                             const float* __restrict__ bias,
                             float* __restrict__ out,
                             int M, int N, int K)
{
    __shared__ _Float16 xh[BM * HS];     // x hi-part, f16   (10 KB)
    __shared__ _Float16 xl[BM * HS];     // x lo-part, f16   (10 KB)
    __shared__ _Float16 wsig[BN * HS];   // sign(W), f16     (20 KB)

    const int tid    = threadIdx.x;
    const int lane   = tid & 31;
    const int wid    = tid >> 5;       // 0..7
    const int wave_m = wid >> 2;       // 0..1  -> 64 rows
    const int wave_n = wid & 3;        // 0..3  -> 64 cols
    const int m0     = blockIdx.y * BM;
    const int n0     = blockIdx.x * BN;
    const int lrow   = lane & 15;
    const int kb     = (lane >> 4) << 3;   // 0 or 8

    v8f acc[4][4];
    #pragma unroll
    for (int mi = 0; mi < 4; ++mi)
        #pragma unroll
        for (int ni = 0; ni < 4; ++ni)
            acc[mi][ni] = (v8f){0.f,0.f,0.f,0.f,0.f,0.f,0.f,0.f};

    // ---- prologue: stage tile 0 into LDS ----
    {
        #pragma unroll
        for (int it = 0; it < 4; ++it) {             // x: 128 rows x 8 float4
            const int idx = tid + it * 256;
            const int row = idx >> 3;
            const int c4  = (idx & 7) << 2;
            v4f xv = *(const v4f*)(x + (size_t)(m0 + row) * K + c4);
            v4h hi, lo;
            #pragma unroll
            for (int j = 0; j < 4; ++j) {
                const float v = xv[j];
                const _Float16 h = (_Float16)v;
                hi[j] = h;
                lo[j] = (_Float16)(v - (float)h);
            }
            *(v4h*)(&xh[row * HS + c4]) = hi;
            *(v4h*)(&xl[row * HS + c4]) = lo;
        }
        #pragma unroll
        for (int it = 0; it < 8; ++it) {             // w: 256 rows x 8 float4
            const int idx = tid + it * 256;
            const int row = idx >> 3;
            const int c4  = (idx & 7) << 2;
            v4f wv = *(const v4f*)(w + (size_t)(n0 + row) * K + c4);
            v4h sg;
            #pragma unroll
            for (int j = 0; j < 4; ++j)
                sg[j] = (_Float16)signf(wv[j]);
            *(v4h*)(&wsig[row * HS + c4]) = sg;
        }
    }
    __syncthreads();

    // ---- main loop: prefetch(k+1) -> compute(k) -> sync -> store -> sync ----
    for (int k0 = 0; k0 < K; k0 += BK) {
        const bool has_next = (k0 + BK) < K;

        v4f xr[4], wr[8];
        if (has_next) {
            #pragma unroll
            for (int it = 0; it < 4; ++it) {
                const int idx = tid + it * 256;
                const int row = idx >> 3;
                const int c4  = (idx & 7) << 2;
                xr[it] = *(const v4f*)(x + (size_t)(m0 + row) * K + (k0 + BK) + c4);
            }
            #pragma unroll
            for (int it = 0; it < 8; ++it) {
                const int idx = tid + it * 256;
                const int row = idx >> 3;
                const int c4  = (idx & 7) << 2;
                wr[it] = *(const v4f*)(w + (size_t)(n0 + row) * K + (k0 + BK) + c4);
            }
        }

        // ---- compute: 32 WMMAs, 24 ds_load_b128 per wave ----
        v16h bf[4];
        #pragma unroll
        for (int ni = 0; ni < 4; ++ni)
            bf[ni] = load_frag(&wsig[(wave_n * 64 + ni * 16 + lrow) * HS + kb]);
        #pragma unroll
        for (int mi = 0; mi < 4; ++mi) {
            const int off = (wave_m * 64 + mi * 16 + lrow) * HS + kb;
            const v16h ah = load_frag(&xh[off]);
            const v16h al = load_frag(&xl[off]);
            #pragma unroll
            for (int ni = 0; ni < 4; ++ni) {
                acc[mi][ni] = __builtin_amdgcn_wmma_f32_16x16x32_f16(
                    false, ah, false, bf[ni], (short)0, acc[mi][ni],
                    false, false);
                acc[mi][ni] = __builtin_amdgcn_wmma_f32_16x16x32_f16(
                    false, al, false, bf[ni], (short)0, acc[mi][ni],
                    false, false);
            }
        }
        __syncthreads();   // everyone done reading before overwrite

        if (has_next) {
            #pragma unroll
            for (int it = 0; it < 4; ++it) {
                const int idx = tid + it * 256;
                const int row = idx >> 3;
                const int c4  = (idx & 7) << 2;
                v4h hi, lo;
                #pragma unroll
                for (int j = 0; j < 4; ++j) {
                    const float v = xr[it][j];
                    const _Float16 h = (_Float16)v;
                    hi[j] = h;
                    lo[j] = (_Float16)(v - (float)h);
                }
                *(v4h*)(&xh[row * HS + c4]) = hi;
                *(v4h*)(&xl[row * HS + c4]) = lo;
            }
            #pragma unroll
            for (int it = 0; it < 8; ++it) {
                const int idx = tid + it * 256;
                const int row = idx >> 3;
                const int c4  = (idx & 7) << 2;
                v4h sg;
                #pragma unroll
                for (int j = 0; j < 4; ++j)
                    sg[j] = (_Float16)signf(wr[it][j]);
                *(v4h*)(&wsig[row * HS + c4]) = sg;
            }
        }
        __syncthreads();   // stores visible before next compute
    }

    // ---- epilogue: + sign(bias), store per f32 C/D layout ----
    // acc element r of lane L -> M = r + 8*(L>=16), N = L%16
    #pragma unroll
    for (int ni = 0; ni < 4; ++ni) {
        const int col = n0 + wave_n * 64 + ni * 16 + lrow;
        const float sb = signf(bias[col]);
        #pragma unroll
        for (int mi = 0; mi < 4; ++mi) {
            const int mbase = m0 + wave_m * 64 + mi * 16 + ((lane >> 4) << 3);
            #pragma unroll
            for (int r = 0; r < 8; ++r)
                out[(size_t)(mbase + r) * N + col] = acc[mi][ni][r] + sb;
        }
    }
}

extern "C" void kernel_launch(void* const* d_in, const int* in_sizes, int n_in,
                              void* d_out, int out_size, void* d_ws, size_t ws_size,
                              hipStream_t stream) {
    const float* x    = (const float*)d_in[0];   // (M, K)
    const float* w    = (const float*)d_in[1];   // (N, K)
    const float* bias = (const float*)d_in[2];   // (N,)
    float* out        = (float*)d_out;           // (M, N)

    const int N = in_sizes[2];
    const int K = in_sizes[1] / N;
    const int M = in_sizes[0] / K;

    dim3 grid(N / BN, M / BM);
    dim3 block(256);
    binlinear_wmma_f16split<<<grid, block, 0, stream>>>(x, w, bias, out, M, N, K);
}
